// SpikeGCL_89687507076355
// MI455X (gfx1250) — compile-verified, matrix operand order checked by reading, not run on hardware
//
#include <hip/hip_runtime.h>

// ---------------- problem constants (match reference) ----------------
#define NN     50000    // nodes
#define NE     800000   // edges
#define TT     32       // timesteps
#define CINF   2048     // input channels
#define CHK    64       // chunk = CIN/T
#define HH     96       // hidden channels
#define VTH    0.005f
#define BN_EPS 1e-5f
#define NB     ((NN + 255) / 256)   // scan blocks

static_assert(NN % 16 == 0, "row tiles must divide N");
static_assert(HH % 16 == 0, "col tiles must divide H");

typedef __attribute__((ext_vector_type(16))) __bf16 v16bf;
typedef __attribute__((ext_vector_type(8)))  float  v8f;

// ---------------- prep kernels ----------------
__global__ void k_zeroi(int* p, int n) {
  int i = blockIdx.x * blockDim.x + threadIdx.x;
  if (i < n) p[i] = 0;
}

__global__ void k_zerof(float* p, int n) {
  int i = blockIdx.x * blockDim.x + threadIdx.x;
  if (i < n) p[i] = 0.0f;
}

// in-degree counts (excluding self loop)
__global__ void k_cnt(const int* __restrict__ ei, int* __restrict__ counts, int ne) {
  int e = blockIdx.x * blockDim.x + threadIdx.x;
  if (e < ne) atomicAdd(&counts[ei[ne + e]], 1);
}

// deg = counts+1 (self loop); dinv = deg^-1/2 ; snorm = 1/deg
__global__ void k_degfin(const int* __restrict__ counts, float* __restrict__ dinv,
                         float* __restrict__ snorm, int n) {
  int i = blockIdx.x * blockDim.x + threadIdx.x;
  if (i < n) {
    float d = (float)(counts[i] + 1);
    dinv[i]  = rsqrtf(d);
    snorm[i] = 1.0f / d;
  }
}

// ---- 3-kernel exclusive scan of counts -> rowStart (CSR offsets) ----
__global__ __launch_bounds__(256) void k_blocksum(const int* __restrict__ counts,
                                                  int* __restrict__ blockSums, int n) {
  __shared__ int sh[256];
  int tid = threadIdx.x;
  int i = blockIdx.x * 256 + tid;
  sh[tid] = (i < n) ? counts[i] : 0;
  __syncthreads();
  for (int off = 128; off > 0; off >>= 1) {
    if (tid < off) sh[tid] += sh[tid + off];
    __syncthreads();
  }
  if (tid == 0) blockSums[blockIdx.x] = sh[0];
}

__global__ void k_scanblocks(int* __restrict__ blockSums, int* __restrict__ rowStart) {
  if (threadIdx.x == 0 && blockIdx.x == 0) {
    int run = 0;
    for (int b = 0; b < NB; ++b) {
      int t = blockSums[b];
      blockSums[b] = run;
      run += t;
    }
    rowStart[NN] = run;   // == NE
  }
}

__global__ __launch_bounds__(256) void k_scanlocal(const int* __restrict__ counts,
                                                   const int* __restrict__ blockSums,
                                                   int* __restrict__ rowStart,
                                                   int* __restrict__ cursor, int n) {
  __shared__ int sh[256];
  int tid = threadIdx.x;
  int i = blockIdx.x * 256 + tid;
  int c = (i < n) ? counts[i] : 0;
  sh[tid] = c;
  __syncthreads();
  // Hillis-Steele inclusive scan
  for (int off = 1; off < 256; off <<= 1) {
    int t = (tid >= off) ? sh[tid - off] : 0;
    __syncthreads();
    sh[tid] += t;
    __syncthreads();
  }
  if (i < n) {
    int rs = sh[tid] - c + blockSums[blockIdx.x];   // exclusive + block offset
    rowStart[i] = rs;
    cursor[i]   = rs;
  }
}

// fill CSR: eSrc[pos] = src node, eW[pos] = dinv[src]*dinv[dst]
__global__ void k_csrfill(const int* __restrict__ ei, const float* __restrict__ dinv,
                          int* __restrict__ cursor, int* __restrict__ eSrc,
                          float* __restrict__ eW, int ne) {
  int e = blockIdx.x * blockDim.x + threadIdx.x;
  if (e >= ne) return;
  int s = ei[e], d = ei[ne + e];
  int pos = atomicAdd(&cursor[d], 1);
  eSrc[pos] = s;
  eW[pos]   = dinv[s] * dinv[d];
}

// fold BN params into scale/shift; reduce lin_w over out-dim
__global__ void k_params(const float* __restrict__ g1, const float* __restrict__ b1,
                         const float* __restrict__ m1, const float* __restrict__ v1,
                         const float* __restrict__ gS, const float* __restrict__ bS,
                         const float* __restrict__ mS, const float* __restrict__ vS,
                         const float* __restrict__ lin_w,
                         float* __restrict__ s1o1,   // [T][128]: scale[64], shift[64]
                         float* __restrict__ sSoS,   // [2*H]
                         float* __restrict__ wsum) { // [H]
  int i = blockIdx.x * blockDim.x + threadIdx.x;
  if (i < TT * CHK) {
    float s = g1[i] * rsqrtf(v1[i] + BN_EPS);
    int t = i / CHK, k = i % CHK;
    s1o1[t * 128 + k]      = s;
    s1o1[t * 128 + 64 + k] = b1[i] - m1[i] * s;
  } else if (i < TT * CHK + HH) {
    int h = i - TT * CHK;
    float s = gS[h] * rsqrtf(vS[h] + BN_EPS);
    sSoS[h]      = s;
    sSoS[HH + h] = bS[h] - mS[h] * s;
  } else if (i < TT * CHK + 2 * HH) {
    int h = i - TT * CHK - HH;
    float acc = 0.0f;
    for (int r = 0; r < HH; ++r) acc += lin_w[r * HH + h];
    wsum[h] = acc;
  }
}

// Build bf16 B-fragments for w1: layout [t][(ct*2+kt)*32+lane][16 elems]
// 16-bit B 32x16: lane -> col n = ct*16 + (lane&15); kbase = kt*32 + 8*(lane>>4)
// element e -> K = kbase + e + (e>=8 ? 8 : 0)
__global__ void k_w1frag(const float* __restrict__ w1, __bf16* __restrict__ wf) {
  int i = blockIdx.x * blockDim.x + threadIdx.x;     // T*12*32 lanes
  if (i >= TT * 12 * 32) return;
  int lane = i & 31;
  int rest = i >> 5;
  int ctkt = rest % 12;
  int t    = rest / 12;
  int ct = ctkt >> 1, kt = ctkt & 1;
  int kbase = kt * 32 + ((lane >> 4) << 3);
  int n = ct * 16 + (lane & 15);
  __bf16* dst = wf + ((size_t)t * 6144 + (size_t)(ctkt * 32 + lane) * 16);
  #pragma unroll
  for (int e = 0; e < 16; ++e) {
    int kk = kbase + e + ((e >= 8) ? 8 : 0);
    dst[e] = (__bf16)w1[((size_t)t * CHK + kk) * HH + n];
  }
}

// Build bf16 B-fragments for w2: layout [(ct*3+kt)*32+lane][16 elems]
__global__ void k_w2frag(const float* __restrict__ w2, __bf16* __restrict__ wf) {
  int i = blockIdx.x * blockDim.x + threadIdx.x;     // 18*32 lanes
  if (i >= 18 * 32) return;
  int lane = i & 31;
  int ctkt = i >> 5;
  int ct = ctkt / 3, kt = ctkt % 3;
  int kbase = kt * 32 + ((lane >> 4) << 3);
  int n = ct * 16 + (lane & 15);
  __bf16* dst = wf + (size_t)(ctkt * 32 + lane) * 16;
  #pragma unroll
  for (int e = 0; e < 16; ++e) {
    int kk = kbase + e + ((e >= 8) ? 8 : 0);
    dst[e] = (__bf16)w2[(size_t)kk * HH + n];
  }
}

// ---------------- GEMM 1: h1 = bn(x_t) @ w1_t  (N x 64 @ 64 x 96) ----------------
// one wave per 16-row tile; 6 column tiles of 16; K = 64 -> 2 steps of 32.
// Fused epilogue: outH = h1 (gather source); outS = h1*snorm[row] + b1_t[col]
__global__ __launch_bounds__(32) void k_gemm1(const float* __restrict__ xt,   // x + t*64, row stride CIN
                                              const float* __restrict__ so,   // s1o1 + t*128
                                              const __bf16* __restrict__ wf,  // w1 frags for t
                                              const float* __restrict__ snorm,
                                              const float* __restrict__ bias, // b1 + t*H
                                              float* __restrict__ outH,
                                              float* __restrict__ outS) {
  int lane    = threadIdx.x;
  int rowBase = blockIdx.x << 4;
  int m       = rowBase + (lane & 15);
  int kb      = (lane >> 4) << 3;
  v8f c[6] = {};
  const float* xr = xt + (size_t)m * CINF;
  #pragma unroll
  for (int kt = 0; kt < 2; ++kt) {
    v16bf a;
    #pragma unroll
    for (int e = 0; e < 16; ++e) {
      int kk = kt * 32 + kb + e + ((e >= 8) ? 8 : 0);
      a[e] = (__bf16)(xr[kk] * so[kk] + so[64 + kk]);
    }
    #pragma unroll
    for (int ct = 0; ct < 6; ++ct) {
      v16bf b = *(const v16bf*)(wf + (size_t)((ct * 2 + kt) * 32 + lane) * 16);
      c[ct] = __builtin_amdgcn_wmma_f32_16x16x32_bf16(false, a, false, b,
                                                      (short)0, c[ct], false, false);
    }
  }
  // C/D layout: vgpr r -> M = rowBase + r + 8*(lane>=16); N = ct*16 + (lane&15)
  int mrow = rowBase + ((lane >> 4) << 3);
  int ncol = lane & 15;
  float sn[8];
  #pragma unroll
  for (int r = 0; r < 8; ++r) sn[r] = snorm[mrow + r];
  float bs[6];
  #pragma unroll
  for (int ct = 0; ct < 6; ++ct) bs[ct] = bias[ct * 16 + ncol];
  #pragma unroll
  for (int ct = 0; ct < 6; ++ct)
    #pragma unroll
    for (int r = 0; r < 8; ++r) {
      size_t idx = (size_t)(mrow + r) * HH + ct * 16 + ncol;
      float cv = c[ct][r];
      outH[idx] = cv;
      outS[idx] = cv * sn[r] + bs[ct];
    }
}

// ---------------- GEMM 2: h2 = bnS(elu(g1)) @ w2  (N x 96 @ 96 x 96) ----------------
// Safe to write outS == g1 buffer: each row is read (A-build) and written
// (epilogue) by exactly the same wave, reads precede writes in program order.
__global__ __launch_bounds__(32) void k_gemm2(const float* __restrict__ g1,   // [N][H]
                                              const float* __restrict__ so,   // sSoS
                                              const __bf16* __restrict__ wf,  // w2 frags
                                              const float* __restrict__ snorm,
                                              const float* __restrict__ bias, // b2
                                              float* __restrict__ outH,
                                              float* __restrict__ outS) {
  int lane    = threadIdx.x;
  int rowBase = blockIdx.x << 4;
  int m       = rowBase + (lane & 15);
  int kb      = (lane >> 4) << 3;
  v8f c[6] = {};
  const float* gr = g1 + (size_t)m * HH;
  #pragma unroll
  for (int kt = 0; kt < 3; ++kt) {
    v16bf a;
    #pragma unroll
    for (int e = 0; e < 16; ++e) {
      int kk = kt * 32 + kb + e + ((e >= 8) ? 8 : 0);
      float val = gr[kk];
      val = (val > 0.0f) ? val : (__expf(val) - 1.0f);           // ELU
      a[e] = (__bf16)(val * so[kk] + so[HH + kk]);               // shared BN
    }
    #pragma unroll
    for (int ct = 0; ct < 6; ++ct) {
      v16bf b = *(const v16bf*)(wf + (size_t)((ct * 3 + kt) * 32 + lane) * 16);
      c[ct] = __builtin_amdgcn_wmma_f32_16x16x32_bf16(false, a, false, b,
                                                      (short)0, c[ct], false, false);
    }
  }
  int mrow = rowBase + ((lane >> 4) << 3);
  int ncol = lane & 15;
  float sn[8];
  #pragma unroll
  for (int r = 0; r < 8; ++r) sn[r] = snorm[mrow + r];
  float bs[6];
  #pragma unroll
  for (int ct = 0; ct < 6; ++ct) bs[ct] = bias[ct * 16 + ncol];
  #pragma unroll
  for (int ct = 0; ct < 6; ++ct)
    #pragma unroll
    for (int r = 0; r < 8; ++r) {
      size_t idx = (size_t)(mrow + r) * HH + ct * 16 + ncol;
      float cv = c[ct][r];
      outH[idx] = cv;
      outS[idx] = cv * sn[r] + bs[ct];
    }
}

// ---------------- pull-mode GCN gather (CSR, no atomics) ----------------
// one wave per dst node, 3 channels/lane; acc[n][c] += sum_e h[src_e][c]*w_e
__global__ __launch_bounds__(256) void k_gather(const int* __restrict__ rowStart,
                                                const int* __restrict__ eSrc,
                                                const float* __restrict__ eW,
                                                const float* __restrict__ h,
                                                float* __restrict__ acc, int nn) {
  int gid  = blockIdx.x * blockDim.x + threadIdx.x;
  int n    = gid >> 5;
  int lane = gid & 31;
  if (n >= nn) return;
  int beg = rowStart[n], end = rowStart[n + 1];
  float a0 = 0.0f, a1 = 0.0f, a2 = 0.0f;
  for (int e = beg; e < end; ++e) {
    int s   = eSrc[e];
    float w = eW[e];
    const float* hs = h + (size_t)s * HH;
    a0 += hs[lane]      * w;
    a1 += hs[lane + 32] * w;
    a2 += hs[lane + 64] * w;
  }
  size_t base = (size_t)n * HH;
  acc[base + lane]      += a0;   // owner-exclusive RMW, no atomics needed
  acc[base + lane + 32] += a1;
  acc[base + lane + 64] += a2;
}

// ---------------- fused gather2 + PLIF + decode ----------------
// g2 row produced in registers; PLIF membrane update, spike, hard reset,
// and decode dot (wave32 shuffle reduce) without materializing g2.
__global__ __launch_bounds__(256) void k_gather_plif(const int* __restrict__ rowStart,
                                                     const int* __restrict__ eSrc,
                                                     const float* __restrict__ eW,
                                                     const float* __restrict__ h,   // h2 (bufA)
                                                     const float* __restrict__ seed,// h2*snorm + b2 (bufB)
                                                     float* __restrict__ v,
                                                     const float* __restrict__ wsum,
                                                     const float* __restrict__ tau,
                                                     float* __restrict__ out, int nn) {
  int gid  = blockIdx.x * blockDim.x + threadIdx.x;
  int n    = gid >> 5;
  int lane = gid & 31;
  if (n >= nn) return;
  int beg = rowStart[n], end = rowStart[n + 1];
  float a0 = 0.0f, a1 = 0.0f, a2 = 0.0f;
  for (int e = beg; e < end; ++e) {
    int s   = eSrc[e];
    float w = eW[e];
    const float* hs = h + (size_t)s * HH;
    a0 += hs[lane]      * w;
    a1 += hs[lane + 32] * w;
    a2 += hs[lane + 64] * w;
  }
  float invtau = 1.0f / tau[0];
  size_t base = (size_t)n * HH;
  float g[3] = {a0, a1, a2};
  float acc = 0.0f;
  #pragma unroll
  for (int j = 0; j < 3; ++j) {
    int hc = lane + 32 * j;
    float dv = seed[base + hc] + g[j];              // g2 = seed + gather
    float vo = v[base + hc];
    float vn = vo + (dv - vo) * invtau;             // membrane update
    float s  = (vn > VTH) ? 1.0f : 0.0f;            // heaviside spike
    v[base + hc] = (1.0f - s) * vn;                 // hard reset
    acc += s * wsum[hc];
  }
  #pragma unroll
  for (int off = 16; off > 0; off >>= 1)
    acc += __shfl_xor(acc, off, 32);
  if (lane == 0) out[n] = acc;
}

// ---------------- host launcher ----------------
extern "C" void kernel_launch(void* const* d_in, const int* in_sizes, int n_in,
                              void* d_out, int out_size, void* d_ws, size_t ws_size,
                              hipStream_t stream) {
  const float* x     = (const float*)d_in[0];
  const int*   ei    = (const int*)  d_in[1];
  const float* w1    = (const float*)d_in[2];
  const float* b1    = (const float*)d_in[3];
  const float* bn1_g = (const float*)d_in[4];
  const float* bn1_b = (const float*)d_in[5];
  const float* bn1_m = (const float*)d_in[6];
  const float* bn1_v = (const float*)d_in[7];
  const float* bns_g = (const float*)d_in[8];
  const float* bns_b = (const float*)d_in[9];
  const float* bns_m = (const float*)d_in[10];
  const float* bns_v = (const float*)d_in[11];
  const float* w2    = (const float*)d_in[12];
  const float* b2    = (const float*)d_in[13];
  const float* lin_w = (const float*)d_in[14];
  const float* tau   = (const float*)d_in[15];
  float* out = (float*)d_out;

  // workspace carve-up (32B aligned blocks)
  char*  base = (char*)d_ws;
  size_t off  = 0;
  auto carve = [&](size_t nbytes) -> char* {
    off = (off + 31) & ~(size_t)31;
    char* p = base + off;
    off += nbytes;
    return p;
  };
  int*   counts    = (int*)  carve(sizeof(int) * NN);
  int*   rowStart  = (int*)  carve(sizeof(int) * (NN + 1));
  int*   cursor    = (int*)  carve(sizeof(int) * NN);
  int*   blockSums = (int*)  carve(sizeof(int) * NB);
  int*   eSrc      = (int*)  carve(sizeof(int) * NE);
  float* eW        = (float*)carve(sizeof(float) * NE);
  float* dinv      = (float*)carve(sizeof(float) * NN);
  float* snorm     = (float*)carve(sizeof(float) * NN);
  float* v         = (float*)carve(sizeof(float) * (size_t)NN * HH);
  float* bufA      = (float*)carve(sizeof(float) * (size_t)NN * HH);
  float* bufB      = (float*)carve(sizeof(float) * (size_t)NN * HH);
  float* s1o1      = (float*)carve(sizeof(float) * TT * 128);
  float* sSoS      = (float*)carve(sizeof(float) * 2 * HH);
  float* wsum      = (float*)carve(sizeof(float) * HH);
  __bf16* w1f      = (__bf16*)carve(sizeof(__bf16) * (size_t)TT * 6144); // T*12*32*16
  __bf16* w2f      = (__bf16*)carve(sizeof(__bf16) * 18 * 32 * 16);
  (void)ws_size; (void)n_in; (void)in_sizes; (void)out_size;

  const int BT = 256;
  int gN   = (NN + BT - 1) / BT;
  int gE   = (NE + BT - 1) / BT;
  int gNH  = ((size_t)NN * HH + BT - 1) / BT;
  int gNw  = ((size_t)NN * 32 + BT - 1) / BT;   // wave-per-node
  int gTiles = NN / 16;

  // ---- prep: degree, norms, CSR build (once, amortized over 64 passes) ----
  k_zeroi<<<gN, BT, 0, stream>>>(counts, NN);
  k_cnt  <<<gE, BT, 0, stream>>>(ei, counts, NE);
  k_degfin<<<gN, BT, 0, stream>>>(counts, dinv, snorm, NN);
  k_blocksum <<<NB, 256, 0, stream>>>(counts, blockSums, NN);
  k_scanblocks<<<1, 32, 0, stream>>>(blockSums, rowStart);
  k_scanlocal<<<NB, 256, 0, stream>>>(counts, blockSums, rowStart, cursor, NN);
  k_csrfill<<<gE, BT, 0, stream>>>(ei, dinv, cursor, eSrc, eW, NE);
  k_zerof<<<gNH, BT, 0, stream>>>(v, (int)((size_t)NN * HH));
  {
    int nwork = TT * CHK + 2 * HH;
    k_params<<<(nwork + BT - 1) / BT, BT, 0, stream>>>(
        bn1_g, bn1_b, bn1_m, bn1_v, bns_g, bns_b, bns_m, bns_v,
        lin_w, s1o1, sSoS, wsum);
  }
  k_w1frag<<<(TT * 12 * 32 + BT - 1) / BT, BT, 0, stream>>>(w1, w1f);
  k_w2frag<<<(18 * 32 + BT - 1) / BT, BT, 0, stream>>>(w2, w2f);

  // ---- timestep loop (sequential PLIF state) ----
  for (int t = 0; t < TT; ++t) {
    // h1 -> bufA ; seed (h1*snorm + b1_t) -> bufB
    k_gemm1<<<gTiles, 32, 0, stream>>>(x + (size_t)t * CHK, s1o1 + t * 128,
                                       w1f + (size_t)t * 6144, snorm,
                                       b1 + (size_t)t * HH, bufA, bufB);
    // bufB += CSR-gather(bufA)  => g1 = gcn(h1) + b1_t   (no atomics)
    k_gather<<<gNw, BT, 0, stream>>>(rowStart, eSrc, eW, bufA, bufB, NN);
    // h2 -> bufA ; seed (h2*snorm + b2) -> bufB (in-place over g1, wave-local safe)
    k_gemm2<<<gTiles, 32, 0, stream>>>(bufB, sSoS, w2f, snorm, b2, bufA, bufB);
    // fused: g2 = seed + CSR-gather(h2); PLIF; decode -> out[t*N ..]
    k_gather_plif<<<gNw, BT, 0, stream>>>(rowStart, eSrc, eW, bufA, bufB,
                                          v, wsum, tau, out + (size_t)t * NN, NN);
  }
}
